// FusedGWLoss_70669391888902
// MI455X (gfx1250) — matrix operand order repaired and checked by reading, not compiled, not run on hardware
//
#include <hip/hip_runtime.h>

typedef _Float16 h16;
typedef __attribute__((ext_vector_type(16))) _Float16 v16h;
typedef __attribute__((ext_vector_type(8)))  _Float16 v8h;
typedef __attribute__((ext_vector_type(8)))  float    v8f;

constexpr int   kN        = 2048;
constexpr int   kD        = 128;
constexpr float kGW       = 20.0f;
constexpr float kInvGamma = 100.0f;           // 1/0.01
constexpr float kScale    = 4194304.0f;       // n1*n2 = 2^22 (exact power of two)
constexpr float kInvScale = 1.0f / 4194304.0f;
constexpr float kLogN     = 7.6246190f;       // log(2048)
constexpr float kLamStep  = 0.01f;

// low 32 bits of a generic pointer into LDS == raw LDS byte address (ISA 10.2)
__device__ __forceinline__ unsigned lds_addr32(const void* p) {
  return (unsigned)(unsigned long long)p;
}

// ---------------------------------------------------------------------------
// WMMA GEMM:  C[M,N] = A[M,K] * B[N,K]^T   (A,B row-major f16, f32 accumulate)
// 128x128 block tile, 256 threads = 8 waves, each wave: 4x2 tiles of 16x16.
// K staged 32-wide through double-buffered LDS via CDNA5 async-to-LDS DMA
// (global_load_async_to_lds_b128, ASYNCcnt), stride 40 halves vs bank conflicts.
// EPI: 0 -> of32 = exp(-g)          (inter_c)
//      1 -> of16 = exp(-g)*adj      (intra_c f16)
//      2 -> of16 = g                (P = s_hat*2^22 @ c2^T)
//      3 -> of32 = g                (M = c1 @ P, still carrying *2^22)
// ---------------------------------------------------------------------------
template <int EPI>
__global__ __launch_bounds__(256) void gw_gemm_abT(
    const h16* __restrict__ A, const h16* __restrict__ B,
    int Kd, int lda, int ldb,
    const float* __restrict__ adj,
    float* __restrict__ of32, h16* __restrict__ of16)
{
  __shared__ alignas(16) h16 lA[2][128 * 40];
  __shared__ alignas(16) h16 lB[2][128 * 40];
  const int tid  = threadIdx.x;
  const int lane = tid & 31;
  const int wv   = tid >> 5;       // 0..7
  const int wm   = wv >> 2;        // 0..1  (64-row strip)
  const int wn   = wv & 3;         // 0..3  (32-col strip)
  const int bm   = blockIdx.y * 128;
  const int bn   = blockIdx.x * 128;
  const int ldr  = tid >> 1;       // row this thread stages (0..127)
  const int ldc  = (tid & 1) * 16; // half-row chunk (0 / 16)

  // per-thread LDS destinations for the async DMA, both buffers
  const unsigned laddrA[2] = { lds_addr32(&lA[0][ldr * 40 + ldc]),
                               lds_addr32(&lA[1][ldr * 40 + ldc]) };
  const unsigned laddrB[2] = { lds_addr32(&lB[0][ldr * 40 + ldc]),
                               lds_addr32(&lB[1][ldr * 40 + ldc]) };
  const h16* gA = A + (size_t)(bm + ldr) * lda + ldc;
  const h16* gB = B + (size_t)(bn + ldr) * ldb + ldc;

  // stage one 32-wide K slab of A and B into LDS buffer `buf` (async, no VGPR bounce)
  auto stage = [&](int buf, int kt) {
    unsigned la = laddrA[buf], lb = laddrB[buf];
    unsigned long long a0 = (unsigned long long)(gA + kt);
    unsigned long long a1 = (unsigned long long)(gA + kt + 8);
    unsigned long long b0 = (unsigned long long)(gB + kt);
    unsigned long long b1 = (unsigned long long)(gB + kt + 8);
    asm volatile(
        "global_load_async_to_lds_b128 %0, %4, off\n\t"
        "global_load_async_to_lds_b128 %1, %5, off\n\t"
        "global_load_async_to_lds_b128 %2, %6, off\n\t"
        "global_load_async_to_lds_b128 %3, %7, off"
        :
        : "v"(la), "v"(la + 16u), "v"(lb), "v"(lb + 16u),
          "v"(a0), "v"(a1), "v"(b0), "v"(b1)
        : "memory");
  };

  v8f acc[4][2];
  v8f vzero;
#pragma unroll
  for (int e = 0; e < 8; ++e) vzero[e] = 0.0f;
#pragma unroll
  for (int i = 0; i < 4; ++i)
#pragma unroll
    for (int j = 0; j < 2; ++j) acc[i][j] = vzero;

  const int nk = Kd / 32;
  stage(0, 0);

  for (int t = 0; t < nk; ++t) {
    // our async slab for buffer t&1 is complete; barrier also guarantees every
    // wave finished *reading* buffer (t+1)&1 (its last wmma forced s_wait_dscnt)
    asm volatile("s_wait_asynccnt 0" ::: "memory");
    __syncthreads();
    if (t + 1 < nk) stage((t + 1) & 1, (t + 1) * 32);

    const h16* bufA = lA[t & 1];
    const h16* bufB = lB[t & 1];
    const int fr  = lane & 15;
    const int akb = (lane & 16) >> 1; // A layout: lanes 16-31 start at K=8
    const int bk0 = (lane & 16) >> 1; // B layout: lanes 16-31 start at K=8

    v16h af[4];
#pragma unroll
    for (int tm = 0; tm < 4; ++tm) {
      const h16* p = &bufA[(wm * 64 + tm * 16 + fr) * 40 + akb];
      v8h lo = *(const v8h*)p;
      v8h hi = *(const v8h*)(p + 16);
#pragma unroll
      for (int e = 0; e < 8; ++e) { af[tm][e] = lo[e]; af[tm][8 + e] = hi[e]; }
    }
    v16h bf[2];
#pragma unroll
    for (int tn = 0; tn < 2; ++tn) {
      const h16* p = &bufB[(wn * 32 + tn * 16 + fr) * 40 + bk0];
      v8h lo = *(const v8h*)p;
      v8h hi = *(const v8h*)(p + 16);
#pragma unroll
      for (int e = 0; e < 8; ++e) { bf[tn][e] = lo[e]; bf[tn][8 + e] = hi[e]; }
    }
#pragma unroll
    for (int tm = 0; tm < 4; ++tm)
#pragma unroll
      for (int tn = 0; tn < 2; ++tn)
        acc[tm][tn] = __builtin_amdgcn_wmma_f32_16x16x32_f16(
            false, af[tm], false, bf[tn], (short)0, acc[tm][tn], false, false);
  }

  // C/D layout: lane holds col = lane&15; VGPR r holds row r (+8 for lanes>=16)
  const int colb = lane & 15;
  const int rb   = (lane >> 4) << 3;
#pragma unroll
  for (int tm = 0; tm < 4; ++tm) {
#pragma unroll
    for (int tn = 0; tn < 2; ++tn) {
      const int col = bn + wn * 32 + tn * 16 + colb;
#pragma unroll
      for (int r = 0; r < 8; ++r) {
        const int row = bm + wm * 64 + tm * 16 + rb + r;
        const size_t idx = (size_t)row * kN + col;
        const float g = acc[tm][tn][r];
        if (EPI == 0)      of32[idx] = __expf(-g);
        else if (EPI == 1) of16[idx] = (h16)(__expf(-g) * adj[idx]);
        else if (EPI == 2) of16[idx] = (h16)g;
        else               of32[idx] = g;
      }
    }
  }
}

// ---------------------------------------------------------------------------
// Small utility kernels
// ---------------------------------------------------------------------------
__global__ void cast_f32_h(const float* __restrict__ in, h16* __restrict__ out, int n) {
  int i = blockIdx.x * blockDim.x + threadIdx.x;
  if (i < n) out[i] = (h16)in[i];
}

__global__ void fill_f32(float* __restrict__ p, float val, size_t n) {
  size_t i = (size_t)blockIdx.x * blockDim.x + threadIdx.x;
  if (i < n) p[i] = val;
}

// out[i] = (s[i] - lam) * 2^22  -> f16  (exact power-of-two rescale)
__global__ void scale_shift_h(const float* __restrict__ s, const float* __restrict__ lam_p,
                              h16* __restrict__ out, size_t n) {
  size_t i = (size_t)blockIdx.x * blockDim.x + threadIdx.x;
  if (i < n) out[i] = (h16)((s[i] - *lam_p) * kScale);
}

__global__ __launch_bounds__(256) void rowsum_f32(const float* __restrict__ mat,
                                                  float* __restrict__ out, int n,
                                                  const float* __restrict__ lam_p,
                                                  float lamscale) {
  __shared__ float red[256];
  const int row = blockIdx.x;
  const float* mr = mat + (size_t)row * n;
  float s = 0.0f;
  for (int j = threadIdx.x; j < n; j += 256) s += mr[j];
  red[threadIdx.x] = s;
  __syncthreads();
  for (int st = 128; st > 0; st >>= 1) {
    if (threadIdx.x < st) red[threadIdx.x] += red[threadIdx.x + st];
    __syncthreads();
  }
  if (threadIdx.x == 0) {
    float l = lam_p ? *lam_p : 0.0f;
    out[row] = red[0] - lamscale * l;
  }
}

__global__ void colsum_f32(const float* __restrict__ mat, float* __restrict__ out, int n,
                           const float* __restrict__ lam_p, float lamscale) {
  int j = blockIdx.x * blockDim.x + threadIdx.x;
  float s = 0.0f;
  for (int i = 0; i < n; ++i) s += mat[(size_t)i * n + j];
  float l = lam_p ? *lam_p : 0.0f;
  out[j] = s - lamscale * l;
}

// rowsum of f16 matrix (optionally of its elementwise square)
__global__ __launch_bounds__(256) void rowsum_h(const h16* __restrict__ mat,
                                                float* __restrict__ out, int n, int square) {
  __shared__ float red[256];
  const int row = blockIdx.x;
  const h16* mr = mat + (size_t)row * n;
  float s = 0.0f;
  for (int j = threadIdx.x; j < n; j += 256) {
    float v = (float)mr[j];
    s += square ? v * v : v;
  }
  red[threadIdx.x] = s;
  __syncthreads();
  for (int st = 128; st > 0; st >>= 1) {
    if (threadIdx.x < st) red[threadIdx.x] += red[threadIdx.x + st];
    __syncthreads();
  }
  if (threadIdx.x == 0) out[row] = red[0];
}

__global__ __launch_bounds__(256) void matvec_f32(const float* __restrict__ mat,
                                                  const float* __restrict__ x,
                                                  float* __restrict__ y, int n) {
  __shared__ float red[256];
  const int row = blockIdx.x;
  const float* mr = mat + (size_t)row * n;
  float s = 0.0f;
  for (int j = threadIdx.x; j < n; j += 256) s += mr[j] * x[j];
  red[threadIdx.x] = s;
  __syncthreads();
  for (int st = 128; st > 0; st >>= 1) {
    if (threadIdx.x < st) red[threadIdx.x] += red[threadIdx.x + st];
    __syncthreads();
  }
  if (threadIdx.x == 0) y[row] = red[0];
}

// y = (mat.^2) @ x  with mat in f16
__global__ __launch_bounds__(256) void matvec_h_sq(const h16* __restrict__ mat,
                                                   const float* __restrict__ x,
                                                   float* __restrict__ y, int n) {
  __shared__ float red[256];
  const int row = blockIdx.x;
  const h16* mr = mat + (size_t)row * n;
  float s = 0.0f;
  for (int j = threadIdx.x; j < n; j += 256) {
    float v = (float)mr[j];
    s += v * v * x[j];
  }
  red[threadIdx.x] = s;
  __syncthreads();
  for (int st = 128; st > 0; st >>= 1) {
    if (threadIdx.x < st) red[threadIdx.x] += red[threadIdx.x + st];
    __syncthreads();
  }
  if (threadIdx.x == 0) y[row] = red[0];
}

// km holds M*2^22 on entry; overwritten with K = -(inter + GW*(r1+r2-2M))/gamma
__global__ void build_K(float* __restrict__ km, const float* __restrict__ inter,
                        const float* __restrict__ r1, const float* __restrict__ r2) {
  size_t idx = (size_t)blockIdx.x * blockDim.x + threadIdx.x;
  int i = (int)(idx >> 11);          // n = 2048
  int j = (int)(idx & 2047);
  float m = km[idx] * kInvScale;
  float gwc = r1[i] + r2[j] - 2.0f * m;
  km[idx] = -(inter[idx] + kGW * gwc) * kInvGamma;
}

// u[row] = log_mu - logsumexp_j(K[row,j] + v[j]);  one wave32 per row
__global__ __launch_bounds__(256) void row_lse(const float* __restrict__ K,
                                               const float* __restrict__ v,
                                               float* __restrict__ u, int n, float log_mu) {
  const int lane = threadIdx.x & 31;
  const int wave = threadIdx.x >> 5;
  const int row  = blockIdx.x * 8 + wave;
  const float* kr = K + (size_t)row * n;
  float m = -1e30f, s = 0.0f;
  for (int j = lane; j < n; j += 32) {
    float x = kr[j] + v[j];
    if (x > m) { s = s * __expf(m - x) + 1.0f; m = x; }
    else       { s += __expf(x - m); }
  }
  for (int off = 16; off > 0; off >>= 1) {
    float mo = __shfl_xor(m, off, 32);
    float so = __shfl_xor(s, off, 32);
    float mn = fmaxf(m, mo);
    s = s * __expf(m - mn) + so * __expf(mo - mn);
    m = mn;
  }
  if (lane == 0) u[row] = log_mu - (m + __logf(s));
}

// column logsumexp in two stages: 16 row-chunks of partial (m,s), then combine
__global__ void col_lse_part(const float* __restrict__ K, const float* __restrict__ u,
                             float* __restrict__ pm, float* __restrict__ ps, int n) {
  const int j = blockIdx.x * blockDim.x + threadIdx.x;
  const int chunk = blockIdx.y;
  const int rows = n >> 4;
  const int i0 = chunk * rows;
  float m = -1e30f, s = 0.0f;
  for (int i = i0; i < i0 + rows; ++i) {
    float x = K[(size_t)i * n + j] + u[i];
    if (x > m) { s = s * __expf(m - x) + 1.0f; m = x; }
    else       { s += __expf(x - m); }
  }
  pm[(size_t)chunk * n + j] = m;
  ps[(size_t)chunk * n + j] = s;
}

__global__ void col_lse_combine(const float* __restrict__ pm, const float* __restrict__ ps,
                                float* __restrict__ v, int n, float log_nu) {
  const int j = blockIdx.x * blockDim.x + threadIdx.x;
  float m = -1e30f, s = 0.0f;
  for (int c = 0; c < 16; ++c) {
    float mo = pm[(size_t)c * n + j];
    float so = ps[(size_t)c * n + j];
    float mn = fmaxf(m, mo);
    s = s * __expf(m - mn) + so * __expf(mo - mn);
    m = mn;
  }
  v[j] = log_nu - (m + __logf(s));
}

__global__ void compute_s(const float* __restrict__ K, const float* __restrict__ u,
                          const float* __restrict__ v, float* __restrict__ s) {
  size_t idx = (size_t)blockIdx.x * blockDim.x + threadIdx.x;
  int i = (int)(idx >> 11);
  int j = (int)(idx & 2047);
  s[idx] = __expf(K[idx] + u[i] + v[j]);
}

// deterministic two-stage full-matrix reduction
// mode 0: sum A ; mode 1: sum A*B ; mode 2: sum A*(B-lam)*scale
__global__ __launch_bounds__(256) void mat_reduce_part(
    const float* __restrict__ A, const float* __restrict__ B,
    const float* __restrict__ lam_p, float scale, int mode,
    float* __restrict__ part, size_t total)
{
  __shared__ float red[256];
  const float lam = lam_p ? *lam_p : 0.0f;
  const size_t stride = (size_t)gridDim.x * blockDim.x;
  float s = 0.0f;
  for (size_t i = (size_t)blockIdx.x * blockDim.x + threadIdx.x; i < total; i += stride) {
    float a = A[i];
    if (mode == 0)      s += a;
    else if (mode == 1) s += a * B[i];
    else                s += a * (B[i] - lam) * scale;
  }
  red[threadIdx.x] = s;
  __syncthreads();
  for (int st = 128; st > 0; st >>= 1) {
    if (threadIdx.x < st) red[threadIdx.x] += red[threadIdx.x + st];
    __syncthreads();
  }
  if (threadIdx.x == 0) part[blockIdx.x] = red[0];
}

// dot (or plain sum when y==nullptr), single block of 1024
__global__ __launch_bounds__(1024) void dot_vec(const float* __restrict__ x,
                                                const float* __restrict__ y,
                                                float* __restrict__ out, int n) {
  __shared__ float red[1024];
  const int t = threadIdx.x;
  float s = 0.0f;
  for (int i = t; i < n; i += 1024) s += x[i] * (y ? y[i] : 1.0f);
  red[t] = s;
  __syncthreads();
  for (int st = 512; st > 0; st >>= 1) {
    if (t < st) red[t] += red[t + st];
    __syncthreads();
  }
  if (t == 0) *out = red[0];
}

__global__ void transpose_h(const h16* __restrict__ in, h16* __restrict__ out, int n) {
  __shared__ h16 tile[32][33];
  int x = blockIdx.x * 32 + threadIdx.x;
  int y0 = blockIdx.y * 32 + threadIdx.y;
  for (int r = 0; r < 32; r += 8) tile[threadIdx.y + r][threadIdx.x] = in[(size_t)(y0 + r) * n + x];
  __syncthreads();
  int x2 = blockIdx.y * 32 + threadIdx.x;
  int y2 = blockIdx.x * 32 + threadIdx.y;
  for (int r = 0; r < 32; r += 8) out[(size_t)(y2 + r) * n + x2] = tile[threadIdx.x][threadIdx.y + r];
}

// sc: 0=k1 1=sum_r1sq 2=sum_c2s 3=sum_rs1 4=sum_rs2 5=d1 6=d2 7=d3
//     8=lam_new 9=sum(inter*s) 10=term1 11=term2 12=term3
__global__ void finalize_lambda(const float* __restrict__ sc, const float* __restrict__ lam0,
                                float* __restrict__ lam_out) {
  const float n = (float)kN;
  float k1 = sc[0];
  float k2 = n * sc[5] + n * sc[6] - 2.0f * sc[7];
  float k3 = n * n * sc[1] + n * n * sc[2] - 2.0f * sc[3] * sc[4];
  float upd = (k1 + 2.0f * kGW * k2) / (2.0f * kGW * k3);
  lam_out[0] = kLamStep * upd + (1.0f - kLamStep) * lam0[0];
}

__global__ void finalize_loss(const float* __restrict__ sc, const float* __restrict__ lam_p,
                              float* __restrict__ loss_out, float* __restrict__ lam_out) {
  float lam = lam_p[0];
  float w_loss  = sc[9] - lam * sc[0];
  float gw_loss = sc[10] + sc[11] - 2.0f * sc[12];
  loss_out[0] = w_loss + kGW * gw_loss + 20.0f;
  lam_out[0]  = lam;
}

// ---------------------------------------------------------------------------
extern "C" void kernel_launch(void* const* d_in, const int* in_sizes, int n_in,
                              void* d_out, int out_size, void* d_ws, size_t ws_size,
                              hipStream_t stream) {
  (void)in_sizes; (void)n_in; (void)out_size; (void)ws_size;
  const float* out1 = (const float*)d_in[0];
  const float* out2 = (const float*)d_in[1];
  const float* adj1 = (const float*)d_in[2];
  const float* adj2 = (const float*)d_in[3];
  const float* lam0 = (const float*)d_in[4];
  float* out = (float*)d_out;

  const size_t NN = (size_t)kN * kN;
  char* w = (char*)d_ws;
  size_t off = 0;
  auto take = [&](size_t bytes) -> char* {
    char* p = w + off;
    off += (bytes + 255) & ~(size_t)255;
    return p;
  };
  float* inter = (float*)take(NN * 4);
  float* sbuf  = (float*)take(NN * 4);
  float* km    = (float*)take(NN * 4);          // holds M, overwritten with K
  h16* h1  = (h16*)take((size_t)kN * kD * 2);
  h16* h2  = (h16*)take((size_t)kN * kD * 2);
  h16* c1h = (h16*)take(NN * 2);
  h16* c2h = (h16*)take(NN * 2);
  h16* Pb  = (h16*)take(NN * 2);
  h16* Pt  = (h16*)take(NN * 2);
  h16* sh  = (h16*)take(NN * 2);
  float* va   = (float*)take(kN * 4);
  float* vb   = (float*)take(kN * 4);
  float* vr1  = (float*)take(kN * 4);
  float* vr2  = (float*)take(kN * 4);
  float* vu   = (float*)take(kN * 4);
  float* vv   = (float*)take(kN * 4);
  float* rs1  = (float*)take(kN * 4);
  float* rs2  = (float*)take(kN * 4);
  float* r1sq = (float*)take(kN * 4);
  float* c2s  = (float*)take(kN * 4);
  float* as_  = (float*)take(kN * 4);
  float* bs_  = (float*)take(kN * 4);
  float* srs2 = (float*)take(kN * 4);
  float* pm   = (float*)take((size_t)16 * kN * 4);
  float* ps   = (float*)take((size_t)16 * kN * 4);
  float* part = (float*)take(1024 * 4);
  float* sc   = (float*)take(64 * 4);

  const dim3 gg(kN / 128, kN / 128);
  const int nflat = (int)(NN / 256);

  // --- cost matrices (f16 WMMA GEMMs with fused epilogues) ---
  cast_f32_h<<<(kN * kD) / 256, 256, 0, stream>>>(out1, h1, kN * kD);
  cast_f32_h<<<(kN * kD) / 256, 256, 0, stream>>>(out2, h2, kN * kD);
  gw_gemm_abT<0><<<gg, 256, 0, stream>>>(h1, h2, kD, kD, kD, nullptr, inter, nullptr);
  gw_gemm_abT<1><<<gg, 256, 0, stream>>>(h1, h1, kD, kD, kD, adj1, nullptr, c1h);
  gw_gemm_abT<1><<<gg, 256, 0, stream>>>(h2, h2, kD, kD, kD, adj2, nullptr, c2h);

  // --- static vectors/scalars for lambda update (c1/c2 are symmetric) ---
  rowsum_h<<<kN, 256, 0, stream>>>(c1h, rs1, kN, 0);
  rowsum_h<<<kN, 256, 0, stream>>>(c2h, rs2, kN, 0);
  rowsum_h<<<kN, 256, 0, stream>>>(c1h, r1sq, kN, 1);
  rowsum_h<<<kN, 256, 0, stream>>>(c2h, c2s, kN, 1);
  dot_vec<<<1, 1024, 0, stream>>>(rs1, nullptr, sc + 3, kN);
  dot_vec<<<1, 1024, 0, stream>>>(rs2, nullptr, sc + 4, kN);
  dot_vec<<<1, 1024, 0, stream>>>(r1sq, nullptr, sc + 1, kN);
  dot_vec<<<1, 1024, 0, stream>>>(c2s, nullptr, sc + 2, kN);
  mat_reduce_part<<<1024, 256, 0, stream>>>(inter, nullptr, nullptr, 1.0f, 0, part, NN);
  dot_vec<<<1, 1024, 0, stream>>>(part, nullptr, sc + 0, 1024);   // k1

  // --- Sinkhorn: s0 uniform ---
  fill_f32<<<nflat, 256, 0, stream>>>(sbuf, kInvScale, NN);

  for (int o = 0; o < 10; ++o) {
    rowsum_f32<<<kN, 256, 0, stream>>>(sbuf, va, kN, lam0, (float)kN);
    colsum_f32<<<kN / 256, 256, 0, stream>>>(sbuf, vb, kN, lam0, (float)kN);
    scale_shift_h<<<nflat, 256, 0, stream>>>(sbuf, lam0, sh, NN);
    matvec_h_sq<<<kN, 256, 0, stream>>>(c1h, va, vr1, kN);
    matvec_h_sq<<<kN, 256, 0, stream>>>(c2h, vb, vr2, kN);
    // M*2^22 = c1 @ ((s_hat*2^22) @ c2^T)
    gw_gemm_abT<2><<<gg, 256, 0, stream>>>(sh, c2h, kN, kN, kN, nullptr, nullptr, Pb);
    transpose_h<<<dim3(kN / 32, kN / 32), dim3(32, 8), 0, stream>>>(Pb, Pt, kN);
    gw_gemm_abT<3><<<gg, 256, 0, stream>>>(c1h, Pt, kN, kN, kN, nullptr, km, nullptr);
    build_K<<<nflat, 256, 0, stream>>>(km, inter, vr1, vr2);
    fill_f32<<<kN / 256, 256, 0, stream>>>(vv, 0.0f, (size_t)kN);
    for (int it = 0; it < 5; ++it) {
      row_lse<<<kN / 8, 256, 0, stream>>>(km, vv, vu, kN, -kLogN);
      col_lse_part<<<dim3(kN / 256, 16), 256, 0, stream>>>(km, vu, pm, ps, kN);
      col_lse_combine<<<kN / 256, 256, 0, stream>>>(pm, ps, vv, kN, -kLogN);
    }
    compute_s<<<nflat, 256, 0, stream>>>(km, vu, vv, sbuf);
  }

  // --- lambda update ---
  rowsum_f32<<<kN, 256, 0, stream>>>(sbuf, as_, kN, nullptr, 0.0f);
  colsum_f32<<<kN / 256, 256, 0, stream>>>(sbuf, bs_, kN, nullptr, 0.0f);
  matvec_f32<<<kN, 256, 0, stream>>>(sbuf, rs2, srs2, kN);
  dot_vec<<<1, 1024, 0, stream>>>(r1sq, as_, sc + 5, kN);
  dot_vec<<<1, 1024, 0, stream>>>(c2s, bs_, sc + 6, kN);
  dot_vec<<<1, 1024, 0, stream>>>(rs1, srs2, sc + 7, kN);
  finalize_lambda<<<1, 1, 0, stream>>>(sc, lam0, sc + 8);

  // --- loss (uses lam_new, which lives on-device at sc+8) ---
  mat_reduce_part<<<1024, 256, 0, stream>>>(inter, sbuf, nullptr, 1.0f, 1, part, NN);
  dot_vec<<<1, 1024, 0, stream>>>(part, nullptr, sc + 9, 1024);
  rowsum_f32<<<kN, 256, 0, stream>>>(sbuf, va, kN, sc + 8, (float)kN);
  colsum_f32<<<kN / 256, 256, 0, stream>>>(sbuf, vb, kN, sc + 8, (float)kN);
  matvec_h_sq<<<kN, 256, 0, stream>>>(c1h, va, vr1, kN);
  matvec_h_sq<<<kN, 256, 0, stream>>>(c2h, vb, vr2, kN);
  dot_vec<<<1, 1024, 0, stream>>>(vr1, va, sc + 10, kN);
  dot_vec<<<1, 1024, 0, stream>>>(vr2, vb, sc + 11, kN);
  scale_shift_h<<<nflat, 256, 0, stream>>>(sbuf, sc + 8, sh, NN);
  gw_gemm_abT<2><<<gg, 256, 0, stream>>>(sh, c2h, kN, kN, kN, nullptr, nullptr, Pb);
  transpose_h<<<dim3(kN / 32, kN / 32), dim3(32, 8), 0, stream>>>(Pb, Pt, kN);
  gw_gemm_abT<3><<<gg, 256, 0, stream>>>(c1h, Pt, kN, kN, kN, nullptr, km, nullptr);
  mat_reduce_part<<<1024, 256, 0, stream>>>(km, sbuf, sc + 8, kInvScale, 2, part, NN);
  dot_vec<<<1, 1024, 0, stream>>>(part, nullptr, sc + 12, 1024);

  finalize_loss<<<1, 1, 0, stream>>>(sc, sc + 8, out, out + 1 + NN);
  hipMemcpyAsync(out + 1, sbuf, NN * sizeof(float), hipMemcpyDeviceToDevice, stream);
}
